// AdvancedFRBDetector_68745246540403
// MI455X (gfx1250) — matrix operand order repaired and checked by reading, not compiled, not run on hardware
//
#include <hip/hip_runtime.h>

typedef __bf16 bf16_t;
typedef __attribute__((ext_vector_type(16))) __bf16 v16bf;
typedef __attribute__((ext_vector_type(8)))  __bf16 v8bf;
typedef __attribute__((ext_vector_type(8)))  float  v8f;

#define BN_INV 0.9999950000374997f   // 1/sqrt(1+1e-5)

// ---------------------------------------------------------------------------
// f32 -> bf16 conversion (weights, once per launch; deterministic)
// ---------------------------------------------------------------------------
__global__ void cvt_f32_bf16_kernel(const float* __restrict__ in, bf16_t* __restrict__ out, int n) {
  int i = blockIdx.x * 256 + threadIdx.x;
  if (i < n) out[i] = (bf16_t)in[i];
}

// ---------------------------------------------------------------------------
// Dedispersion shift table: shifts[d][chunk] = trunc(mean over 128 f of delay)
// ---------------------------------------------------------------------------
__global__ void shifts_kernel(const float* __restrict__ dmv, int* __restrict__ shifts) {
  int tid = threadIdx.x;           // 256 threads = 32 d * 8 chunks
  int d = tid >> 3, c = tid & 7;
  float dm = dmv[d];
  float s = 0.f;
  for (int j = 0; j < 128; j++) {
    int f = c * 128 + j;
    float fs = (float)f / 1023.f + 0.1f;
    fs = 1.f / (fs * fs);
    s += tanhf(dm * fs * 1e-3f) * 500.f;
  }
  shifts[d * 8 + c] = (int)truncf(s * (1.f / 128.f));
}

// ---------------------------------------------------------------------------
// Pooled DM features: pooled[bd][ti*32+fi] = mean over 64x32 block of
// x[b, (t+shift)%2048, f]  (ded never materialized; x lives in L2)
// ---------------------------------------------------------------------------
__global__ void pool_kernel(const float* __restrict__ x, const int* __restrict__ shifts,
                            bf16_t* __restrict__ pooled) {
  int blk = blockIdx.x;                 // bd*1024 + ti*32 + fi
  int fi = blk & 31, ti = (blk >> 5) & 31, bd = blk >> 10;
  int b = bd >> 5, d = bd & 31;
  __shared__ float red[256];
  float s = 0.f;
  for (int i = 0; i < 8; i++) {
    int e = threadIdx.x + 256 * i;      // 2048 elements per block
    int tt = e >> 5, ff = e & 31;
    int t = ti * 64 + tt, f = fi * 32 + ff;
    int st = (t + shifts[d * 8 + (f >> 7)]) & 2047;
    s += x[((size_t)b * 2048 + st) * 1024 + f];
  }
  red[threadIdx.x] = s; __syncthreads();
  for (int o = 128; o > 0; o >>= 1) { if (threadIdx.x < o) red[threadIdx.x] += red[threadIdx.x + o]; __syncthreads(); }
  if (threadIdx.x == 0) pooled[bd * 1024 + ti * 32 + fi] = (bf16_t)(red[0] * (1.f / 2048.f));
}

// ---------------------------------------------------------------------------
// Row softmax over 32 columns (one wave per row)
// ---------------------------------------------------------------------------
__global__ void softmax32_kernel(const float* __restrict__ in, float* __restrict__ out) {
  int r = blockIdx.x, l = threadIdx.x;
  float v = in[r * 32 + l];
  float m = v;
  for (int o = 16; o > 0; o >>= 1) m = fmaxf(m, __shfl_xor(m, o, 32));
  float e = __expf(v - m);
  float su = e;
  for (int o = 16; o > 0; o >>= 1) su += __shfl_xor(su, o, 32);
  out[r * 32 + l] = e / su;
}

// ---------------------------------------------------------------------------
// dm_corr[b,t,f] = sum_d attn[b,d] * x[b,(t+shift[d,f/128])%T,f]   (L2-bound)
// ---------------------------------------------------------------------------
__global__ void dmcorr_kernel(const float* __restrict__ x, const int* __restrict__ shifts,
                              const float* __restrict__ attn, float* __restrict__ out) {
  size_t idx = (size_t)blockIdx.x * 256 + threadIdx.x;   // 2*2048*1024
  int f = idx & 1023;
  int t = (int)((idx >> 10) & 2047);
  int b = (int)(idx >> 21);
  int fc = f >> 7;
  float s = 0.f;
  for (int d = 0; d < 32; d++) {
    int st = (t + shifts[d * 8 + fc]) & 2047;
    s += attn[b * 32 + d] * x[((size_t)b * 2048 + st) * 1024 + f];
  }
  out[idx] = s;
}

// ---------------------------------------------------------------------------
// Direct 3x3 conv, pad 1, fused bias + BN(eval) + ReLU  (frontend ~10 GFLOP)
// ---------------------------------------------------------------------------
__global__ void conv3x3_bn_relu(const float* __restrict__ in, const float* __restrict__ w,
                                const float* __restrict__ bias, const float* __restrict__ g,
                                const float* __restrict__ beta, float* __restrict__ out,
                                int Bn, int Cin, int H, int Wd, int Cout, int Hout, int Wout,
                                int sh, int sw) {
  size_t idx = (size_t)blockIdx.x * 256 + threadIdx.x;
  size_t total = (size_t)Bn * Cout * Hout * Wout;
  if (idx >= total) return;
  int ow = (int)(idx % Wout);
  int oh = (int)((idx / Wout) % Hout);
  int oc = (int)((idx / ((size_t)Wout * Hout)) % Cout);
  int b  = (int)(idx / ((size_t)Wout * Hout * Cout));
  int ih0 = oh * sh - 1, iw0 = ow * sw - 1;
  float s = 0.f;
  for (int ic = 0; ic < Cin; ic++) {
    const float* ip = in + ((size_t)(b * Cin + ic) * H) * Wd;
    const float* wp = w + (size_t)(oc * Cin + ic) * 9;
#pragma unroll
    for (int kh = 0; kh < 3; kh++) {
      int ih = ih0 + kh; if (ih < 0 || ih >= H) continue;
#pragma unroll
      for (int kw = 0; kw < 3; kw++) {
        int iw = iw0 + kw; if (iw < 0 || iw >= Wd) continue;
        s += ip[(size_t)ih * Wd + iw] * wp[kh * 3 + kw];
      }
    }
  }
  s += bias[oc];
  s = s * BN_INV * g[oc] + beta[oc];
  out[idx] = fmaxf(s, 0.f);
}

// ---------------------------------------------------------------------------
// Pack conv4 output (b,c,t,f) -> bf16 A [(b*128+t), c*16+f] for proj GEMM
// ---------------------------------------------------------------------------
__global__ void pack_proj_A(const float* __restrict__ c3, bf16_t* __restrict__ A) {
  int idx = blockIdx.x * 256 + threadIdx.x;   // 2^20 elements
  int f = idx & 15, t = (idx >> 4) & 127, c = (idx >> 11) & 255, b = idx >> 19;
  A[((size_t)(b * 128 + t)) * 4096 + c * 16 + f] = (bf16_t)c3[idx];
}

// ---------------------------------------------------------------------------
// Generic bf16 WMMA GEMM:  out = act(A[M,K] @ W[N,K]^T + bias) * resScale + res
// One wave computes one 16x16 tile; K stepped by 32 via v_wmma_f32_16x16x32_bf16.
// Loads are clamped (no divergence) so EXEC is all-ones at the WMMA.
// ---------------------------------------------------------------------------
__global__ void gemm_bf16_wmma(const bf16_t* __restrict__ A, const bf16_t* __restrict__ W,
                               const float* __restrict__ bias,
                               float* __restrict__ outF, bf16_t* __restrict__ outB,
                               const float* __restrict__ residual, float resScale,
                               int M, int N, int K, int act) {
  const int lane = threadIdx.x & 31;
  const int tn = blockIdx.x, tm = blockIdx.y;
  const int half = lane >> 4;
  const int l16  = lane & 15;
  const int arow = tm * 16 + l16;
  const int bcol = tn * 16 + l16;
  const int koffA = half ? 8 : 0;     // A: lanes 16-31 hold K 8..15 / 24..31
  const int koffB = half ? 8 : 0;     // B: lanes 16-31 hold K 8..15 / 24..31
  const bf16_t* ap = A + (size_t)((arow < M) ? arow : 0) * K;
  const bf16_t* wp = W + (size_t)((bcol < N) ? bcol : 0) * K;
  v8f acc = {0.f, 0.f, 0.f, 0.f, 0.f, 0.f, 0.f, 0.f};
  for (int k0 = 0; k0 < K; k0 += 32) {
    __builtin_prefetch((const void*)(ap + k0 + 128), 0, 1);   // global_prefetch_b8
    __builtin_prefetch((const void*)(wp + k0 + 128), 0, 1);
    v8bf a0 = *(const v8bf*)(ap + k0 + koffA);
    v8bf a1 = *(const v8bf*)(ap + k0 + 16 + koffA);
    v8bf b0 = *(const v8bf*)(wp + k0 + koffB);
    v8bf b1 = *(const v8bf*)(wp + k0 + 16 + koffB);
    v16bf av, bv;
#pragma unroll
    for (int i = 0; i < 8; i++) { av[i] = a0[i]; av[8 + i] = a1[i]; bv[i] = b0[i]; bv[8 + i] = b1[i]; }
    acc = __builtin_amdgcn_wmma_f32_16x16x32_bf16(false, av, false, bv, (short)0, acc, false, false);
  }
  float bb = 0.f;
  if (bias && bcol < N) bb = bias[bcol];
#pragma unroll
  for (int r = 0; r < 8; r++) {
    int row = tm * 16 + half * 8 + r;   // C/D: lanes 0-15 rows 0-7, lanes 16-31 rows 8-15
    if (row < M && bcol < N) {
      float v = acc[r] + bb;
      if (act == 1)      v = fmaxf(v, 0.f);
      else if (act == 2) v = v / (1.f + __expf(-v));   // swish
      v *= resScale;
      if (residual) v += residual[(size_t)row * N + bcol];
      if (outF) outF[(size_t)row * N + bcol] = v;
      if (outB) outB[(size_t)row * N + bcol] = (bf16_t)v;
    }
  }
}

// ---------------------------------------------------------------------------
// LayerNorm over 256 channels; optional f32 and/or bf16 output
// ---------------------------------------------------------------------------
__global__ void layernorm_kernel(const float* __restrict__ in, const float* __restrict__ g,
                                 const float* __restrict__ b, float* __restrict__ outF,
                                 bf16_t* __restrict__ outB) {
  int row = blockIdx.x, c = threadIdx.x;   // 256 threads
  __shared__ float red[256];
  float v = in[(size_t)row * 256 + c];
  red[c] = v; __syncthreads();
  for (int o = 128; o > 0; o >>= 1) { if (c < o) red[c] += red[c + o]; __syncthreads(); }
  float mean = red[0] * (1.f / 256.f); __syncthreads();
  float dv = v - mean;
  red[c] = dv * dv; __syncthreads();
  for (int o = 128; o > 0; o >>= 1) { if (c < o) red[c] += red[c + o]; __syncthreads(); }
  float y = dv * rsqrtf(red[0] * (1.f / 256.f) + 1e-5f) * g[c] + b[c];
  if (outF) outF[(size_t)row * 256 + c] = y;
  if (outB) outB[(size_t)row * 256 + c] = (bf16_t)y;
}

// ---------------------------------------------------------------------------
// MHSA core (n=128, h=4, dh=64): one block per (b,head,query)
// ---------------------------------------------------------------------------
__global__ void mhsa_kernel(const float* __restrict__ qkv, bf16_t* __restrict__ o) {
  int blk = blockIdx.x;
  int q = blk & 127, hh = (blk >> 7) & 3, b = blk >> 9;
  int j = threadIdx.x;                  // 128 threads
  __shared__ float qs[64];
  __shared__ float p[128];
  __shared__ float r2[128];
  if (j < 64) qs[j] = qkv[((size_t)(b * 128 + q)) * 768 + hh * 64 + j];
  __syncthreads();
  const float* krow = qkv + ((size_t)(b * 128 + j)) * 768 + 256 + hh * 64;
  float s = 0.f;
  for (int d2 = 0; d2 < 64; d2++) s += qs[d2] * krow[d2];
  s *= 0.125f;                          // dh^-0.5
  r2[j] = s; __syncthreads();
  for (int o2 = 64; o2 > 0; o2 >>= 1) { if (j < o2) r2[j] = fmaxf(r2[j], r2[j + o2]); __syncthreads(); }
  float mx = r2[0]; __syncthreads();
  float e = __expf(s - mx);
  r2[j] = e; __syncthreads();
  for (int o2 = 64; o2 > 0; o2 >>= 1) { if (j < o2) r2[j] += r2[j + o2]; __syncthreads(); }
  p[j] = e / r2[0]; __syncthreads();
  if (j < 64) {
    float acc = 0.f;
    for (int kk = 0; kk < 128; kk++)
      acc += p[kk] * qkv[((size_t)(b * 128 + kk)) * 768 + 512 + hh * 64 + j];
    o[((size_t)(b * 128 + q)) * 256 + hh * 64 + j] = (bf16_t)acc;
  }
}

// GLU: z[row,c] = y[row,c] * sigmoid(y[row,256+c])
__global__ void glu_kernel(const float* __restrict__ y, float* __restrict__ z) {
  int idx = blockIdx.x * 256 + threadIdx.x;   // 256*256
  int row = idx >> 8, c = idx & 255;
  float a = y[(size_t)row * 512 + c];
  float gte = y[(size_t)row * 512 + 256 + c];
  z[idx] = a * (1.f / (1.f + __expf(-gte)));
}

// Depthwise 15-tap conv over time + BN(eval) + swish -> bf16
__global__ void dwconv_bn_swish(const float* __restrict__ z, const float* __restrict__ w,
                                const float* __restrict__ bias, const float* __restrict__ g,
                                const float* __restrict__ beta, bf16_t* __restrict__ out) {
  int idx = blockIdx.x * 256 + threadIdx.x;   // 2*128*256
  int c = idx & 255, t = (idx >> 8) & 127, b = idx >> 15;
  float s = 0.f;
#pragma unroll
  for (int k = 0; k < 15; k++) {
    int tt = t + k - 7;
    if (tt >= 0 && tt < 128) s += z[((size_t)(b * 128 + tt)) * 256 + c] * w[c * 15 + k];
  }
  s += bias[c];
  s = s * BN_INV * g[c] + beta[c];
  s = s / (1.f + __expf(-s));
  out[idx] = (bf16_t)s;
}

// gfeat = mean over t ; comb = concat(gfeat, attn) in bf16
__global__ void mean_concat_kernel(const float* __restrict__ h, const float* __restrict__ attn,
                                   bf16_t* __restrict__ comb) {
  int b = blockIdx.x, c = threadIdx.x;   // 288 threads
  if (c < 256) {
    float s = 0.f;
    for (int t = 0; t < 128; t++) s += h[((size_t)(b * 128 + t)) * 256 + c];
    comb[b * 288 + c] = (bf16_t)(s * (1.f / 128.f));
  } else if (c < 288) {
    comb[b * 288 + c] = (bf16_t)attn[b * 32 + (c - 256)];
  }
}

// Final 2-way classifier head
__global__ void cls2_kernel(const float* __restrict__ h1, const float* __restrict__ w,
                            const float* __restrict__ b, float* __restrict__ out) {
  int i = threadIdx.x;
  if (i < 4) {
    int bb = i >> 1, j = i & 1;
    float s = b[j];
    for (int k = 0; k < 128; k++) s += h1[bb * 128 + k] * w[j * 128 + k];
    out[bb * 2 + j] = s;
  }
}

// ---------------------------------------------------------------------------
static inline void launch_gemm(hipStream_t s, const bf16_t* A, const bf16_t* W, const float* bias,
                               float* outF, bf16_t* outB, const float* res, float scale,
                               int M, int N, int K, int act) {
  dim3 grid((N + 15) / 16, (M + 15) / 16);
  gemm_bf16_wmma<<<grid, 32, 0, s>>>(A, W, bias, outF, outB, res, scale, M, N, K, act);
}

extern "C" void kernel_launch(void* const* d_in, const int* in_sizes, int n_in,
                              void* d_out, int out_size, void* d_ws, size_t ws_size,
                              hipStream_t stream) {
  (void)in_sizes; (void)n_in; (void)out_size; (void)ws_size;
  const float* x   = (const float*)d_in[0];
  const float* dmv = (const float*)d_in[1];

  char* wsb = (char*)d_ws;
  size_t off = 0;
  auto alloc = [&](size_t bytes) -> void* {
    void* p = wsb + off;
    off += (bytes + 255) & ~((size_t)255);
    return p;
  };
  auto cvtw = [&](int idx, size_t n) -> bf16_t* {
    bf16_t* dst = (bf16_t*)alloc(n * sizeof(bf16_t));
    cvt_f32_bf16_kernel<<<(unsigned)((n + 255) / 256), 256, 0, stream>>>((const float*)d_in[idx], dst, (int)n);
    return dst;
  };

  // ---- weights -> bf16 (input index map per setup_inputs order) ----
  bf16_t* Wdmf1 = cvtw(2, 128 * 1024);
  bf16_t* Wdmf2 = cvtw(4, 32 * 128);
  bf16_t* Wdma1 = cvtw(6, 256 * 1024);
  bf16_t* Wdma2 = cvtw(8, 64 * 256);
  bf16_t* Wdma3 = cvtw(10, 32 * 64);
  bf16_t* Wproj = cvtw(28, 256 * 4096);
  bf16_t *Wff1a[3], *Wff1b[3], *Wff2a[3], *Wff2b[3], *Wqkv[3], *Wao[3], *Wpw1[3], *Wpw2[3];
  for (int i = 0; i < 3; i++) {
    int cb = 30 + 31 * i;
    Wff1a[i] = cvtw(cb + 10, 1024 * 256);
    Wff1b[i] = cvtw(cb + 12, 256 * 1024);
    Wff2a[i] = cvtw(cb + 14, 1024 * 256);
    Wff2b[i] = cvtw(cb + 16, 256 * 1024);
    Wqkv[i]  = cvtw(cb + 18, 768 * 256);
    Wao[i]   = cvtw(cb + 19, 256 * 256);
    Wpw1[i]  = cvtw(cb + 23, 512 * 256);
    Wpw2[i]  = cvtw(cb + 29, 256 * 256);
  }
  bf16_t* Wcls1 = cvtw(123, 128 * 288);

  // ---- activation / scratch buffers ----
  int*    shifts = (int*)alloc(32 * 8 * 4);
  float*  logits = (float*)alloc(2 * 32 * 4);
  float*  attnW  = (float*)alloc(2 * 32 * 4);
  bf16_t* pooled = (bf16_t*)alloc((size_t)64 * 1024 * 2);
  bf16_t* feat1  = (bf16_t*)alloc((size_t)64 * 128 * 2);
  bf16_t* feat2  = (bf16_t*)alloc((size_t)64 * 32 * 2);   // == [2,1024] a-input
  bf16_t* a1     = (bf16_t*)alloc((size_t)2 * 256 * 2);
  bf16_t* a2     = (bf16_t*)alloc((size_t)2 * 64 * 2);
  float*  dmcorr = (float*)alloc((size_t)2 * 2048 * 1024 * 4);
  float*  c0     = (float*)alloc((size_t)2 * 32 * 1024 * 512 * 4);
  float*  c1     = (float*)alloc((size_t)2 * 64 * 512 * 256 * 4);
  float*  c2     = (float*)alloc((size_t)2 * 128 * 256 * 64 * 4);
  float*  c3     = (float*)alloc((size_t)2 * 256 * 128 * 16 * 4);
  bf16_t* Aproj  = (bf16_t*)alloc((size_t)256 * 4096 * 2);
  float*  h      = (float*)alloc((size_t)256 * 256 * 4);
  float*  tmpF   = (float*)alloc((size_t)256 * 256 * 4);
  bf16_t* tbf    = (bf16_t*)alloc((size_t)256 * 256 * 2);
  bf16_t* ubf    = (bf16_t*)alloc((size_t)256 * 1024 * 2);
  float*  qkv    = (float*)alloc((size_t)256 * 768 * 4);
  bf16_t* obf    = (bf16_t*)alloc((size_t)256 * 256 * 2);
  float*  y1     = (float*)alloc((size_t)256 * 512 * 4);
  float*  z      = (float*)alloc((size_t)256 * 256 * 4);
  bf16_t* dwbf   = (bf16_t*)alloc((size_t)256 * 256 * 2);
  bf16_t* comb   = (bf16_t*)alloc((size_t)2 * 288 * 2);
  float*  h1     = (float*)alloc((size_t)2 * 128 * 4);

  // ---- dedispersion shifts, pooled DM features ----
  shifts_kernel<<<1, 256, 0, stream>>>(dmv, shifts);
  pool_kernel<<<65536, 256, 0, stream>>>(x, shifts, pooled);

  // ---- DM feature + attention MLPs (WMMA) ----
  launch_gemm(stream, pooled, Wdmf1, (const float*)d_in[3],  nullptr, feat1, nullptr, 1.f, 64, 128, 1024, 1);
  launch_gemm(stream, feat1,  Wdmf2, (const float*)d_in[5],  nullptr, feat2, nullptr, 1.f, 64, 32, 128, 1);
  launch_gemm(stream, feat2,  Wdma1, (const float*)d_in[7],  nullptr, a1,    nullptr, 1.f, 2, 256, 1024, 1);
  launch_gemm(stream, a1,     Wdma2, (const float*)d_in[9],  nullptr, a2,    nullptr, 1.f, 2, 64, 256, 1);
  launch_gemm(stream, a2,     Wdma3, (const float*)d_in[11], logits,  nullptr, nullptr, 1.f, 2, 32, 64, 0);
  softmax32_kernel<<<2, 32, 0, stream>>>(logits, attnW);

  // ---- attention-weighted dedispersed sum (L2-bound gather) ----
  dmcorr_kernel<<<16384, 256, 0, stream>>>(x, shifts, attnW, dmcorr);

  // ---- conv frontend ----
  conv3x3_bn_relu<<<131072, 256, 0, stream>>>(dmcorr, (const float*)d_in[12], (const float*)d_in[13],
      (const float*)d_in[14], (const float*)d_in[15], c0, 2, 1, 2048, 1024, 32, 1024, 512, 2, 2);
  conv3x3_bn_relu<<<65536, 256, 0, stream>>>(c0, (const float*)d_in[16], (const float*)d_in[17],
      (const float*)d_in[18], (const float*)d_in[19], c1, 2, 32, 1024, 512, 64, 512, 256, 2, 2);
  conv3x3_bn_relu<<<16384, 256, 0, stream>>>(c1, (const float*)d_in[20], (const float*)d_in[21],
      (const float*)d_in[22], (const float*)d_in[23], c2, 2, 64, 512, 256, 128, 256, 64, 2, 4);
  conv3x3_bn_relu<<<4096, 256, 0, stream>>>(c2, (const float*)d_in[24], (const float*)d_in[25],
      (const float*)d_in[26], (const float*)d_in[27], c3, 2, 128, 256, 64, 256, 128, 16, 2, 4);

  // ---- projection into conformer space ----
  pack_proj_A<<<4096, 256, 0, stream>>>(c3, Aproj);
  launch_gemm(stream, Aproj, Wproj, (const float*)d_in[29], h, nullptr, nullptr, 1.f, 256, 256, 4096, 0);

  // ---- 3 conformer blocks ----
  for (int i = 0; i < 3; i++) {
    int cb = 30 + 31 * i;
    // ff1: h += 0.5 * l2(swish(l1(LN(h))))
    layernorm_kernel<<<256, 256, 0, stream>>>(h, (const float*)d_in[cb + 0], (const float*)d_in[cb + 1], nullptr, tbf);
    launch_gemm(stream, tbf, Wff1a[i], (const float*)d_in[cb + 11], nullptr, ubf, nullptr, 1.f, 256, 1024, 256, 2);
    launch_gemm(stream, ubf, Wff1b[i], (const float*)d_in[cb + 13], h, nullptr, h, 0.5f, 256, 256, 1024, 0);
    // mhsa: h += out(attn(LN(h)))
    layernorm_kernel<<<256, 256, 0, stream>>>(h, (const float*)d_in[cb + 2], (const float*)d_in[cb + 3], nullptr, tbf);
    launch_gemm(stream, tbf, Wqkv[i], nullptr, qkv, nullptr, nullptr, 1.f, 256, 768, 256, 0);
    mhsa_kernel<<<1024, 128, 0, stream>>>(qkv, obf);
    launch_gemm(stream, obf, Wao[i], (const float*)d_in[cb + 20], h, nullptr, h, 1.f, 256, 256, 256, 0);
    // conv module: h += pw2(swish(bn(dw(glu(pw1(LN(LN(h,norm3),conv.ln)))))))
    layernorm_kernel<<<256, 256, 0, stream>>>(h, (const float*)d_in[cb + 4], (const float*)d_in[cb + 5], tmpF, nullptr);
    layernorm_kernel<<<256, 256, 0, stream>>>(tmpF, (const float*)d_in[cb + 21], (const float*)d_in[cb + 22], nullptr, tbf);
    launch_gemm(stream, tbf, Wpw1[i], (const float*)d_in[cb + 24], y1, nullptr, nullptr, 1.f, 256, 512, 256, 0);
    glu_kernel<<<256, 256, 0, stream>>>(y1, z);
    dwconv_bn_swish<<<256, 256, 0, stream>>>(z, (const float*)d_in[cb + 25], (const float*)d_in[cb + 26],
                                             (const float*)d_in[cb + 27], (const float*)d_in[cb + 28], dwbf);
    launch_gemm(stream, dwbf, Wpw2[i], (const float*)d_in[cb + 30], h, nullptr, h, 1.f, 256, 256, 256, 0);
    // ff2: h += 0.5 * l2(swish(l1(LN(h))))
    layernorm_kernel<<<256, 256, 0, stream>>>(h, (const float*)d_in[cb + 6], (const float*)d_in[cb + 7], nullptr, tbf);
    launch_gemm(stream, tbf, Wff2a[i], (const float*)d_in[cb + 15], nullptr, ubf, nullptr, 1.f, 256, 1024, 256, 2);
    launch_gemm(stream, ubf, Wff2b[i], (const float*)d_in[cb + 17], h, nullptr, h, 0.5f, 256, 256, 1024, 0);
    // final LN (in place)
    layernorm_kernel<<<256, 256, 0, stream>>>(h, (const float*)d_in[cb + 8], (const float*)d_in[cb + 9], h, nullptr);
  }

  // ---- classifier ----
  mean_concat_kernel<<<2, 288, 0, stream>>>(h, attnW, comb);
  launch_gemm(stream, comb, Wcls1, (const float*)d_in[124], h1, nullptr, nullptr, 1.f, 2, 128, 288, 1);
  cls2_kernel<<<1, 32, 0, stream>>>(h1, (const float*)d_in[125], (const float*)d_in[126], (float*)d_out);
}
